// SimpleTwoBranchModel_86474871538190
// MI455X (gfx1250) — matrix operand order repaired and checked by reading, not compiled, not run on hardware
//
#include <hip/hip_runtime.h>
#include <hip/hip_bf16.h>
#include <math.h>

typedef __attribute__((ext_vector_type(16))) _Float16 v16h;
typedef __attribute__((ext_vector_type(8)))  _Float16 v8h;
typedef __attribute__((ext_vector_type(8)))  float    v8f;
typedef __attribute__((ext_vector_type(4)))  float    v4f;

#define B_    64
#define T_    1024
#define D_    512
#define H_    512
#define O_    128
#define HALF_ 256

#define TM   128
#define TN   64
#define TK   32
#define PAD  8
#define KSTR (TK + PAD)   // 40 halves (80 bytes) per LDS row: 16B-aligned, bank-spread

#define NC   16           // scan chunks over T
#define CL   (T_ / NC)    // chunk length = 64
#define BH   (B_ * H_)    // 32768

#define ALPHA_M 0.8f

// Build a v16h WMMA fragment from two 16-byte LDS reads.
__device__ __forceinline__ v16h frag_from(const _Float16* row, int off0, int off1) {
  v8h lo = *(const v8h*)(row + off0);
  v8h hi = *(const v8h*)(row + off1);
  v16h r;
#pragma unroll
  for (int i = 0; i < 8; ++i) { r[i] = lo[i]; r[i + 8] = hi[i]; }
  return r;
}

// ---------------------------------------------------------------------------
// Phase 1: d{1,2}_in[t,b,h] = (f16) ( x[b,t, z*HALF + :HALF] . w{1,2}[h,:] + b{1,2}[h] )
// A = x rows (M = B*T), B = w (stored [h][k] == [N][K], K-contiguous).
// ---------------------------------------------------------------------------
__global__ __launch_bounds__(256) void gemm_in_kernel(
    const float* __restrict__ x,  const float* __restrict__ w1,
    const float* __restrict__ w2, const float* __restrict__ b1,
    const float* __restrict__ b2, _Float16* __restrict__ dout1,
    _Float16* __restrict__ dout2) {
  __shared__ alignas(16) _Float16 As[TM][KSTR];
  __shared__ alignas(16) _Float16 Bs[TN][KSTR];

  const int z        = blockIdx.z;
  const float* w     = z ? w2 : w1;
  const float* bias  = z ? b2 : b1;
  _Float16* dout     = z ? dout2 : dout1;
  const int xcol     = z * HALF_;

  const int tid  = threadIdx.x;
  const int lane = tid & 31;
  const int wave = tid >> 5;
  const int lrow = lane & 15;
  const int hsel = lane >> 4;          // which half of the wave
  const int wm   = (wave >> 1) * 32;   // 4x2 wave grid over 128x64 tile
  const int wn   = (wave & 1) * 32;

  const int tile_m = blockIdx.x * TM;  // over M = B*T
  const int tile_n = blockIdx.y * TN;  // over H

  v8f acc[2][2] = {};

  for (int kk = 0; kk < HALF_; kk += TK) {
    // Stage A: 128 rows x 32 f32 -> f16 LDS. 1024 float4 chunks, 4/thread.
#pragma unroll
    for (int it = 0; it < 4; ++it) {
      int lin = tid + it * 256;
      int row = lin >> 3;
      int c4  = (lin & 7) * 4;
      const float* src = x + (size_t)(tile_m + row) * D_ + xcol + kk + c4;
      if (kk + TK < HALF_) __builtin_prefetch(src + TK, 0, 3);
      v4f v = *(const v4f*)src;
#pragma unroll
      for (int j = 0; j < 4; ++j) As[row][c4 + j] = (_Float16)v[j];
    }
    // Stage B: 64 rows x 32 f32 -> f16 LDS. 512 chunks, 2/thread.
#pragma unroll
    for (int it = 0; it < 2; ++it) {
      int lin = tid + it * 256;
      int row = lin >> 3;
      int c4  = (lin & 7) * 4;
      v4f v = *(const v4f*)(w + (size_t)(tile_n + row) * HALF_ + kk + c4);
#pragma unroll
      for (int j = 0; j < 4; ++j) Bs[row][c4 + j] = (_Float16)v[j];
    }
    __syncthreads();

    v16h afrag[2], bfrag[2];
#pragma unroll
    for (int mt = 0; mt < 2; ++mt) {
      // A 16x32 f16 layout: lane l -> row l%16; j<8: k=8*hsel+j ; j>=8: k=8*hsel+16+(j-8)
      const _Float16* ar = &As[wm + mt * 16 + lrow][0];
      afrag[mt] = frag_from(ar, 8 * hsel, 8 * hsel + 16);
    }
#pragma unroll
    for (int nt = 0; nt < 2; ++nt) {
      // B 32x16 f16 layout: lane l -> col l%16; j<8: k=8*hsel+j ; j>=8: k=8*hsel+16+(j-8)
      const _Float16* br = &Bs[wn + nt * 16 + lrow][0];
      bfrag[nt] = frag_from(br, 8 * hsel, 8 * hsel + 16);
    }
#pragma unroll
    for (int mt = 0; mt < 2; ++mt)
#pragma unroll
      for (int nt = 0; nt < 2; ++nt)
        acc[mt][nt] = __builtin_amdgcn_wmma_f32_16x16x32_f16(
            false, afrag[mt], false, bfrag[nt], (short)0, acc[mt][nt],
            false, false);
    __syncthreads();
  }

  // Epilogue: C/D layout -> m = r + 8*hsel, n = lrow. Scatter to [T,B,H] f16.
#pragma unroll
  for (int mt = 0; mt < 2; ++mt) {
#pragma unroll
    for (int nt = 0; nt < 2; ++nt) {
      int gn   = tile_n + wn + nt * 16 + lrow;
      float bn = bias[gn];
#pragma unroll
      for (int r = 0; r < 8; ++r) {
        int gm = tile_m + wm + mt * 16 + 8 * hsel + r;   // = b*T + t
        int b  = gm >> 10;
        int t  = gm & (T_ - 1);
        size_t off = (size_t)t * BH + (size_t)b * H_ + gn;
        dout[off] = (_Float16)(acc[mt][nt][r] + bn);
      }
    }
  }
}

// ---------------------------------------------------------------------------
// Phase 2: blocked parallel scan of the linear double-EMA recurrence.
//   d1[t] = a1 d1[t-1] + (1-a1) i1[t]   (same for d2)
//   m[t]  = am m[t-1] + (1-am)(d1[t] + d2[t])
// Chunk transition is affine in (d1_in, d2_in, m_in):
//   d1_out = a1^L d1_in + d1e
//   m_out  = am^L m_in + c1 d1_in + c2 d2_in + me,  c(t)=am*c(t-1)+(1-am)*a^{t+1}
// ---------------------------------------------------------------------------

// Pass A: per-chunk local scan from zero state; emit chunk-end states.
__global__ __launch_bounds__(256) void scan_chunk_end_kernel(
    const _Float16* __restrict__ d1in, const _Float16* __restrict__ d2in,
    const float* __restrict__ tau1, const float* __restrict__ tau2,
    float* __restrict__ e1, float* __restrict__ e2, float* __restrict__ em) {
  const int gid  = blockIdx.x * 256 + threadIdx.x;   // 0 .. NC*BH/8 - 1
  const int c    = gid >> 12;                        // / (BH/8=4096)
  const int grp  = gid & 4095;
  const int base = grp * 8;                          // flat b*H + h0
  const int h0   = base & (H_ - 1);

  float a1[8], a2[8], na1[8], na2[8];
#pragma unroll
  for (int j = 0; j < 8; ++j) {
    a1[j]  = 1.0f / (1.0f + __expf(-tau1[h0 + j]));
    a2[j]  = 1.0f / (1.0f + __expf(-tau2[h0 + j]));
    na1[j] = 1.0f - a1[j];
    na2[j] = 1.0f - a2[j];
  }
  float d1[8] = {}, d2[8] = {}, m[8] = {};

  const int t0 = c * CL;
  for (int t = t0; t < t0 + CL; ++t) {
    size_t off = (size_t)t * BH + base;
    v8h i1 = *(const v8h*)(d1in + off);
    v8h i2 = *(const v8h*)(d2in + off);
#pragma unroll
    for (int j = 0; j < 8; ++j) {
      d1[j] = fmaf(a1[j], d1[j], na1[j] * (float)i1[j]);
      d2[j] = fmaf(a2[j], d2[j], na2[j] * (float)i2[j]);
      m[j]  = fmaf(ALPHA_M, m[j], (1.0f - ALPHA_M) * (d1[j] + d2[j]));
    }
  }
  const size_t so = (size_t)c * BH + base;
#pragma unroll
  for (int j = 0; j < 8; ++j) {
    e1[so + j] = d1[j];
    e2[so + j] = d2[j];
    em[so + j] = m[j];
  }
}

// Pass B: compose the NC chunk transitions serially (per h-column) to get
// each chunk's true starting state. One thread per (b,h).
__global__ __launch_bounds__(256) void scan_combine_kernel(
    const float* __restrict__ tau1, const float* __restrict__ tau2,
    const float* __restrict__ e1, const float* __restrict__ e2,
    const float* __restrict__ em,
    float* __restrict__ s1, float* __restrict__ s2, float* __restrict__ sm) {
  const int bh = blockIdx.x * 256 + threadIdx.x;     // 0 .. BH-1
  const int h  = bh & (H_ - 1);

  const float a1 = 1.0f / (1.0f + __expf(-tau1[h]));
  const float a2 = 1.0f / (1.0f + __expf(-tau2[h]));

  // Per-chunk propagation scalars: a1^L, a2^L, am^L, c1(L-1), c2(L-1).
  float p1 = 1.0f, p2 = 1.0f, pm = 1.0f, c1 = 0.0f, c2 = 0.0f;
  for (int t = 0; t < CL; ++t) {
    p1 *= a1;
    p2 *= a2;
    pm *= ALPHA_M;
    c1 = fmaf(ALPHA_M, c1, (1.0f - ALPHA_M) * p1);
    c2 = fmaf(ALPHA_M, c2, (1.0f - ALPHA_M) * p2);
  }

  float sd1 = 0.0f, sd2 = 0.0f, smm = 0.0f;
#pragma unroll
  for (int c = 0; c < NC; ++c) {
    size_t off = (size_t)c * BH + bh;
    s1[off] = sd1;
    s2[off] = sd2;
    sm[off] = smm;
    float nd1 = fmaf(p1, sd1, e1[off]);
    float nd2 = fmaf(p2, sd2, e2[off]);
    float nm  = fmaf(pm, smm, fmaf(c1, sd1, fmaf(c2, sd2, em[off])));
    sd1 = nd1; sd2 = nd2; smm = nm;
  }
}

// Pass C: per-chunk scan from the true starting state; write mems (f16).
__global__ __launch_bounds__(256) void scan_write_kernel(
    const _Float16* __restrict__ d1in, const _Float16* __restrict__ d2in,
    const float* __restrict__ tau1, const float* __restrict__ tau2,
    const float* __restrict__ s1, const float* __restrict__ s2,
    const float* __restrict__ sm, _Float16* __restrict__ mems) {
  const int gid  = blockIdx.x * 256 + threadIdx.x;
  const int c    = gid >> 12;
  const int grp  = gid & 4095;
  const int base = grp * 8;
  const int h0   = base & (H_ - 1);

  float a1[8], a2[8], na1[8], na2[8];
#pragma unroll
  for (int j = 0; j < 8; ++j) {
    a1[j]  = 1.0f / (1.0f + __expf(-tau1[h0 + j]));
    a2[j]  = 1.0f / (1.0f + __expf(-tau2[h0 + j]));
    na1[j] = 1.0f - a1[j];
    na2[j] = 1.0f - a2[j];
  }
  float d1[8], d2[8], m[8];
  const size_t so = (size_t)c * BH + base;
#pragma unroll
  for (int j = 0; j < 8; ++j) {
    d1[j] = s1[so + j];
    d2[j] = s2[so + j];
    m[j]  = sm[so + j];
  }

  const int t0 = c * CL;
  for (int t = t0; t < t0 + CL; ++t) {
    size_t off = (size_t)t * BH + base;
    v8h i1 = *(const v8h*)(d1in + off);
    v8h i2 = *(const v8h*)(d2in + off);
    v8h o;
#pragma unroll
    for (int j = 0; j < 8; ++j) {
      d1[j] = fmaf(a1[j], d1[j], na1[j] * (float)i1[j]);
      d2[j] = fmaf(a2[j], d2[j], na2[j] * (float)i2[j]);
      m[j]  = fmaf(ALPHA_M, m[j], (1.0f - ALPHA_M) * (d1[j] + d2[j]));
      o[j]  = (_Float16)m[j];
    }
    *(v8h*)(mems + off) = o;
  }
}

// ---------------------------------------------------------------------------
// Phase 3: out[b,t,o] = sigmoid( mems[t,b,:] . wo[o,:] + bo[o] )
// A = mems rows (M = T*B, already f16), B = wo ([O][H] = [N][K]).
// ---------------------------------------------------------------------------
__global__ __launch_bounds__(256) void gemm_out_kernel(
    const _Float16* __restrict__ mems, const float* __restrict__ wo,
    const float* __restrict__ bo, float* __restrict__ out) {
  __shared__ alignas(16) _Float16 As[TM][KSTR];
  __shared__ alignas(16) _Float16 Bs[TN][KSTR];

  const int tid  = threadIdx.x;
  const int lane = tid & 31;
  const int wave = tid >> 5;
  const int lrow = lane & 15;
  const int hsel = lane >> 4;
  const int wm   = (wave >> 1) * 32;
  const int wn   = (wave & 1) * 32;

  const int tile_m = blockIdx.x * TM;  // over M = T*B
  const int tile_n = blockIdx.y * TN;  // over O

  v8f acc[2][2] = {};

  for (int kk = 0; kk < H_; kk += TK) {
    // Stage A: f16 direct, 128 rows x 32 halves = 512 b128 chunks, 2/thread.
#pragma unroll
    for (int it = 0; it < 2; ++it) {
      int lin = tid + it * 256;
      int row = lin >> 2;
      int c8  = (lin & 3) * 8;
      const _Float16* src = mems + (size_t)(tile_m + row) * H_ + kk + c8;
      if (kk + TK < H_) __builtin_prefetch(src + TK, 0, 3);
      v8h v = *(const v8h*)src;
      *(v8h*)&As[row][c8] = v;
    }
    // Stage B: wo f32 -> f16, 64 rows x 32, 2/thread.
#pragma unroll
    for (int it = 0; it < 2; ++it) {
      int lin = tid + it * 256;
      int row = lin >> 3;
      int c4  = (lin & 7) * 4;
      v4f v = *(const v4f*)(wo + (size_t)(tile_n + row) * H_ + kk + c4);
#pragma unroll
      for (int j = 0; j < 4; ++j) Bs[row][c4 + j] = (_Float16)v[j];
    }
    __syncthreads();

    v16h afrag[2], bfrag[2];
#pragma unroll
    for (int mt = 0; mt < 2; ++mt) {
      const _Float16* ar = &As[wm + mt * 16 + lrow][0];
      afrag[mt] = frag_from(ar, 8 * hsel, 8 * hsel + 16);
    }
#pragma unroll
    for (int nt = 0; nt < 2; ++nt) {
      const _Float16* br = &Bs[wn + nt * 16 + lrow][0];
      bfrag[nt] = frag_from(br, 8 * hsel, 8 * hsel + 16);
    }
#pragma unroll
    for (int mt = 0; mt < 2; ++mt)
#pragma unroll
      for (int nt = 0; nt < 2; ++nt)
        acc[mt][nt] = __builtin_amdgcn_wmma_f32_16x16x32_f16(
            false, afrag[mt], false, bfrag[nt], (short)0, acc[mt][nt],
            false, false);
    __syncthreads();
  }

  // Epilogue: bias + sigmoid, scatter to out[b,t,o]. Row gm = t*B + b.
#pragma unroll
  for (int mt = 0; mt < 2; ++mt) {
#pragma unroll
    for (int nt = 0; nt < 2; ++nt) {
      int gn   = tile_n + wn + nt * 16 + lrow;
      float bn = bo[gn];
#pragma unroll
      for (int r = 0; r < 8; ++r) {
        int gm = tile_m + wm + mt * 16 + 8 * hsel + r;
        int t  = gm >> 6;          // / B_
        int b  = gm & (B_ - 1);
        size_t off = (size_t)b * (T_ * O_) + (size_t)t * O_ + gn;
        float v = acc[mt][nt][r] + bn;
        out[off] = 1.0f / (1.0f + __expf(-v));
      }
    }
  }
}

// ---------------------------------------------------------------------------
extern "C" void kernel_launch(void* const* d_in, const int* in_sizes, int n_in,
                              void* d_out, int out_size, void* d_ws, size_t ws_size,
                              hipStream_t stream) {
  const float* x    = (const float*)d_in[0];
  const float* w1   = (const float*)d_in[1];
  const float* b1   = (const float*)d_in[2];
  const float* tau1 = (const float*)d_in[3];
  const float* w2   = (const float*)d_in[4];
  const float* b2   = (const float*)d_in[5];
  const float* tau2 = (const float*)d_in[6];
  const float* wo   = (const float*)d_in[7];
  const float* bo   = (const float*)d_in[8];
  float* out = (float*)d_out;

  const size_t n  = (size_t)T_ * BH;              // 33.5M elements
  const size_t ns = (size_t)NC * BH;              // 524288 elements
  _Float16* d1 = (_Float16*)d_ws;                 // 64 MiB
  _Float16* d2 = d1 + n;                          // 64 MiB
  _Float16* mm = d2 + n;                          // 64 MiB
  float* e1 = (float*)(mm + n);                   // 6 x 2 MiB of scan states
  float* e2 = e1 + ns;
  float* em = e2 + ns;
  float* s1 = em + ns;
  float* s2 = s1 + ns;
  float* sm = s2 + ns;

  dim3 g1((B_ * T_) / TM, H_ / TN, 2);            // (512, 8, 2)
  gemm_in_kernel<<<g1, 256, 0, stream>>>(x, w1, w2, b1, b2, d1, d2);

  const int scanBlocks = (NC * BH / 8) / 256;     // 256
  scan_chunk_end_kernel<<<scanBlocks, 256, 0, stream>>>(d1, d2, tau1, tau2,
                                                        e1, e2, em);
  scan_combine_kernel<<<BH / 256, 256, 0, stream>>>(tau1, tau2, e1, e2, em,
                                                    s1, s2, sm);
  scan_write_kernel<<<scanBlocks, 256, 0, stream>>>(d1, d2, tau1, tau2,
                                                    s1, s2, sm, mm);

  dim3 g3((T_ * B_) / TM, O_ / TN);               // (512, 2)
  gemm_out_kernel<<<g3, 256, 0, stream>>>(mm, wo, bo, out);
}
